// WaveletAttention_55559696941624
// MI455X (gfx1250) — matrix operand order repaired and checked by reading, not compiled, not run on hardware
//
#include <hip/hip_runtime.h>
#include <hip/hip_bf16.h>
#include <math.h>

typedef __attribute__((ext_vector_type(16))) _Float16 v16h;
typedef __attribute__((ext_vector_type(8)))  float    v8f;

// db8 decomposition low-pass filter (dec_lo), 16 taps
__constant__ float c_db8[16] = {
    -0.00011747678400228192f,  0.0006754494059985568f, -0.0003917403729959771f,
    -0.00487035299301066f,     0.008746094047015655f,   0.013981027917015516f,
    -0.04408825393106472f,    -0.01736930100202211f,    0.128747426620186f,
     0.00047248457399797254f, -0.2840155429624281f,    -0.015829105256023893f,
     0.5853546836548691f,      0.6756307362980128f,     0.3128715909144659f,
     0.05441584224308161f};

// weight of extended-signal position p in the sum over all 135 stride-2 outputs:
//   wp(p) = sum over taps j (same parity as p, (p-j)/2 in [0,134]) of filt_rev[j]
__device__ __forceinline__ float wp_weight(int p) {
  float s = 0.f;
  #pragma unroll
  for (int j = 0; j <= 15; ++j) {
    int pm = p - j;
    if (((pm & 1) == 0) && pm >= 0 && pm <= 268) s += c_db8[15 - j];  // filt_rev[j]
  }
  return s;
}

// -------- Pass 1: y[b,c] = (1/135^2) * sum_{i,j} u[i] u[j] x[b,c,i,j] --------
__global__ __launch_bounds__(256) void wav_reduce(const float* __restrict__ x,
                                                  float* __restrict__ y) {
  __shared__ float uS[256];
  __shared__ float red[256];
  const int t = threadIdx.x;

  // effective 1D weight u[t] (symmetric pad, drop-first, stride-2, 16 taps)
  float uv = wp_weight(t + 14);            // main span
  if (t <= 13)  uv += wp_weight(13 - t);   // left reflection
  if (t >= 241) uv += wp_weight(525 - t);  // right reflection
  uS[t] = uv;
  __syncthreads();

  const int tx = t & 63, ty = t >> 6;      // 64 x 4 tile of threads
  const size_t plane = blockIdx.x;         // b*256 + c
  const float4* xp = (const float4*)(x + plane * 65536);
  const float4  uc = ((const float4*)uS)[tx];

  float acc = 0.f;
  for (int r = ty; r < 256; r += 4) {
    float4 v = xp[r * 64 + tx];            // 128-bit coalesced loads
    acc += uS[r] * (uc.x * v.x + uc.y * v.y + uc.z * v.z + uc.w * v.w);
  }
  red[t] = acc;
  __syncthreads();
  for (int s = 128; s > 0; s >>= 1) {
    if (t < s) red[t] += red[t + s];
    __syncthreads();
  }
  if (t == 0) y[plane] = red[0] * (1.0f / (135.0f * 135.0f));
}

// -------- Pass 2: w = sigmoid(relu(y @ W1^T) @ W2^T), via WMMA --------------
__global__ __launch_bounds__(512) void wav_mlp(const float* __restrict__ y,
                                               const float* __restrict__ W1,
                                               const float* __restrict__ W2,
                                               float* __restrict__ scale) {
  __shared__ _Float16 yS[16 * 256];  // batch padded 8 -> 16
  __shared__ _Float16 hS[16 * 32];   // hidden, K padded 16 -> 32 with zeros
  const int t = threadIdx.x;

  for (int i = t; i < 16 * 256; i += 512) {
    int b = i >> 8, c = i & 255;
    yS[i] = (b < 8) ? (_Float16)y[b * 256 + c] : (_Float16)0.f;
  }
  __syncthreads();

  const int lane  = t & 31;
  const int wave  = t >> 5;
  const int mrow  = lane & 15;
  const int kbA   = (lane < 16) ? 0 : 8;    // 16-bit A 16x32 lane-group K split
  const int kbB   = (lane < 16) ? 0 : 16;   // lane-group K split, contiguous k = 16h + e, used for A and B
  const int nlane = lane & 15;
  const int mOut  = (lane < 16) ? 0 : 8;    // f32 C/D layout

  // GEMM1: [16,256] x [256,16] -> [16,16], wave 0 only (EXEC all-ones in-wave)
  if (wave == 0) {
    v8f acc = {};
    #pragma unroll
    for (int k0 = 0; k0 < 256; k0 += 32) {
      v16h a, b;
      #pragma unroll
      for (int e = 0; e < 16; ++e) {
        int ka = k0 + kbB + e;
        a[e] = yS[mrow * 256 + ka];
        int kb = k0 + kbB + e;
        b[e] = (_Float16)W1[nlane * 256 + kb];  // B[k][n] = W1[n][k]
      }
      acc = __builtin_amdgcn_wmma_f32_16x16x32_f16(
          false, a, false, b, (short)0, acc, false, false);
    }
    #pragma unroll
    for (int v = 0; v < 8; ++v) {
      float r = acc[v] > 0.f ? acc[v] : 0.f;          // ReLU
      hS[(mOut + v) * 32 + nlane]      = (_Float16)r;
      hS[(mOut + v) * 32 + 16 + nlane] = (_Float16)0.f;  // K zero-pad
    }
  }
  __syncthreads();

  // GEMM2: [16,32pad] x [32pad,16] per wave -> 16 waves cover N=256
  const int n0 = wave << 4;
  v16h a2, b2;
  #pragma unroll
  for (int e = 0; e < 16; ++e) {
    int ka = kbB + e;
    a2[e] = hS[mrow * 32 + ka];
    int kb = kbB + e;
    b2[e] = (kb < 16) ? (_Float16)W2[(n0 + nlane) * 16 + kb] : (_Float16)0.f;
  }
  v8f acc2 = {};
  acc2 = __builtin_amdgcn_wmma_f32_16x16x32_f16(
      false, a2, false, b2, (short)0, acc2, false, false);

  #pragma unroll
  for (int v = 0; v < 8; ++v) {
    int m = mOut + v;
    if (m < 8) {  // only real batch rows
      float s = 1.0f / (1.0f + expf(-acc2[v]));       // sigmoid
      scale[m * 256 + n0 + nlane] = s;
    }
  }
}

// -------- Pass 3: out = x * w[b,c]  (pure HBM stream, float4) ---------------
__global__ __launch_bounds__(256) void wav_scale(const float* __restrict__ x,
                                                 const float* __restrict__ scale,
                                                 float* __restrict__ out) {
  const size_t plane = blockIdx.x;         // b*256 + c
  const float  s = scale[plane];
  const float4* xi = (const float4*)(x + plane * 65536);
  float4*       xo = (float4*)(out + plane * 65536);
  const int base = blockIdx.y * 1024;      // 16 chunks of 1024 float4 per plane
  #pragma unroll
  for (int i = 0; i < 4; ++i) {
    int idx = base + i * 256 + threadIdx.x;
    float4 v = xi[idx];
    v.x *= s; v.y *= s; v.z *= s; v.w *= s;
    xo[idx] = v;
  }
}

extern "C" void kernel_launch(void* const* d_in, const int* in_sizes, int n_in,
                              void* d_out, int out_size, void* d_ws, size_t ws_size,
                              hipStream_t stream) {
  const float* x  = (const float*)d_in[0];   // [8,256,256,256] f32
  const float* W1 = (const float*)d_in[1];   // [16,256] f32
  const float* W2 = (const float*)d_in[2];   // [256,16] f32
  float* out = (float*)d_out;                // [8,256,256,256] f32

  float* y_ws  = (float*)d_ws;               // [8*256] pooled LL means
  float* sc_ws = (float*)d_ws + 2048;        // [8*256] sigmoid gates

  wav_reduce<<<dim3(2048), dim3(256), 0, stream>>>(x, y_ws);
  wav_mlp<<<dim3(1), dim3(512), 0, stream>>>(y_ws, W1, W2, sc_ws);
  wav_scale<<<dim3(2048, 16), dim3(256), 0, stream>>>(x, sc_ws, out);
}
